// BaseGNN_87368224735342
// MI455X (gfx1250) — matrix operand order repaired and checked by reading, not compiled, not run on hardware
//
#include <hip/hip_runtime.h>

// ---------------------------------------------------------------------------
// GraphCast-style GNN for MI455X (gfx1250), wave32 + v_wmma_f32_16x16x32_f16.
// All hidden-dim (64) matmuls run on WMMA with f16 activations / fp32 accum.
// Scatter-sum uses fp32 global atomics. 16 rows per wave, 8 waves per block.
// ---------------------------------------------------------------------------

typedef __attribute__((ext_vector_type(16))) _Float16 v16h;
typedef __attribute__((ext_vector_type(8)))  _Float16 v8h;
typedef __attribute__((ext_vector_type(8)))  float    v8f;

#define NGRID 131072
#define NMESH 16384
#define NEDGE 400000
#define NB    2

__device__ __forceinline__ v8f wmma16(v16h a, v16h b, v8f c) {
  // D = A(16x32 f16) * B(32x16 f16) + C(16x16 f32)
  return __builtin_amdgcn_wmma_f32_16x16x32_f16(false, a, false, b, (short)0, c,
                                                false, false);
}

// A fragment (16x32, f16): lane = row (lane&15); per ISA layout, lane half hi
// holds K = {k0+8*hi .. +7} and {k0+16+8*hi .. +7}.
__device__ __forceinline__ v16h a_frag_h(const _Float16* row, int k0, int hi) {
  v8h lo = *(const v8h*)(row + k0 + 8 * hi);
  v8h hh = *(const v8h*)(row + k0 + 16 + 8 * hi);
  v16h a;
#pragma unroll
  for (int i = 0; i < 8; ++i) { a[i] = lo[i]; a[8 + i] = hh[i]; }
  return a;
}

// Same, but source rows are fp32 (aggregate buffers) converted on load.
__device__ __forceinline__ v16h a_frag_f(const float* row, int k0, int hi) {
  const float* p = row + k0 + 8 * hi;
  const float* q = row + k0 + 16 + 8 * hi;
  v16h a;
#pragma unroll
  for (int i = 0; i < 8; ++i) { a[i] = (_Float16)p[i]; a[8 + i] = (_Float16)q[i]; }
  return a;
}

// B fragment: pre-packed so each lane reads its 16 halves contiguously.
__device__ __forceinline__ v16h b_frag(const _Float16* Wp, int f, int lane) {
  return *(const v16h*)(Wp + (((long)f * 32 + lane) << 4));
}

__device__ __forceinline__ float silu_f(float v) { return v / (1.f + __expf(-v)); }

// ---------------------------------------------------------------------------
// Pack fp32 weight W (K x Ncols, row major, K % 32 == 0) into B fragments.
// Fragment f = kc*NT + nt; element (lane, j) = W[kc*32 + (j&7) + 8*(lane>>4) + 16*(j>>3)]
//                                              [nt*16 + (lane&15)]  (0 if pad)
// ---------------------------------------------------------------------------
__global__ void pack_w_kernel(const float* __restrict__ W, _Float16* __restrict__ dst,
                              int K, int Ncols) {
  int NT = (Ncols + 15) >> 4;
  int total = (K >> 5) * NT * 512;
  int idx = blockIdx.x * 256 + threadIdx.x;
  if (idx >= total) return;
  int j = idx & 15;
  int lane = (idx >> 4) & 31;
  int f = idx >> 9;
  int nt = f % NT;
  int kc = f / NT;
  int n = nt * 16 + (lane & 15);
  int k = kc * 32 + (j & 7) + 8 * (lane >> 4) + 16 * (j >> 3);
  float v = 0.f;
  if (n < Ncols && k < K) v = W[(long)k * Ncols + n];
  dst[idx] = (_Float16)v;
}

// ---------------------------------------------------------------------------
// Small-K first layers (scalar): h = silu(x @ W0 + b0), f16 output.
// ---------------------------------------------------------------------------
__global__ void grid_l1_kernel(const float* __restrict__ prev,
                               const float* __restrict__ W0,   // 10 x 64
                               const float* __restrict__ b0,
                               _Float16* __restrict__ outH) {
  long idx = (long)blockIdx.x * 256 + threadIdx.x;   // B*NGRID*64 threads
  int n = (int)(idx & 63);
  long row = idx >> 6;                               // b*NGRID + node
  int b = (int)(row >> 17);
  long node = row & (NGRID - 1);
  float acc = b0[n];
#pragma unroll
  for (int f = 0; f < 10; ++f)
    acc += prev[((long)b * 10 + f) * NGRID + node] * W0[f * 64 + n];
  outH[row * 64 + n] = (_Float16)silu_f(acc);
}

__global__ void feat3_l1_kernel(const float* __restrict__ feats,  // (R,3)
                                const float* __restrict__ W0,     // 3 x 64
                                const float* __restrict__ b0,
                                _Float16* __restrict__ outH) {
  long idx = (long)blockIdx.x * 256 + threadIdx.x;   // R*64 threads
  int n = (int)(idx & 63);
  long row = idx >> 6;
  float acc = b0[n];
#pragma unroll
  for (int f = 0; f < 3; ++f)
    acc += feats[row * 3 + f] * W0[f * 64 + n];
  outH[row * 64 + n] = (_Float16)silu_f(acc);
}

// ---------------------------------------------------------------------------
// Generic Y = f(X @ W + bias) with f16 X (R,64), K = 64.
// flags bit0 = SiLU, bit1 = LayerNorm(g,bb). Optional residual (f16) added
// after LN. grid.x = R / 128 (8 waves x 16 rows).
// ---------------------------------------------------------------------------
__global__ __launch_bounds__(256) void gemm64_kernel(
    const _Float16* __restrict__ X, const _Float16* __restrict__ Wp,
    const float* __restrict__ bias, const float* __restrict__ lng,
    const float* __restrict__ lnb, const _Float16* __restrict__ residual,
    _Float16* __restrict__ Y, int flags) {
  __shared__ float sh[8][16][64];
  __shared__ float smu[8][16];
  __shared__ float srs[8][16];
  int lane = threadIdx.x & 31, wave = threadIdx.x >> 5, hi = lane >> 4;
  long row0 = ((long)blockIdx.x * 8 + wave) * 16;
  const _Float16* rowp = X + (row0 + (lane & 15)) * 64;
  v16h a0 = a_frag_h(rowp, 0, hi);
  v16h a1 = a_frag_h(rowp, 32, hi);
  int n0 = lane & 15;
#pragma unroll
  for (int nt = 0; nt < 4; ++nt) {
    v8f c = {};
    c = wmma16(a0, b_frag(Wp, nt, lane), c);
    c = wmma16(a1, b_frag(Wp, 4 + nt, lane), c);
    float bv = bias[nt * 16 + n0];
#pragma unroll
    for (int r = 0; r < 8; ++r) sh[wave][r + 8 * hi][nt * 16 + n0] = c[r] + bv;
  }
  __syncthreads();
  if (flags & 2) {
    if (lane < 16) {
      float s = 0.f, s2 = 0.f;
      for (int n = 0; n < 64; ++n) { float v = sh[wave][lane][n]; s += v; s2 += v * v; }
      float mu = s * 0.015625f;
      smu[wave][lane] = mu;
      srs[wave][lane] = rsqrtf(fmaxf(s2 * 0.015625f - mu * mu, 0.f) + 1e-5f);
    }
  }
  __syncthreads();
  for (int idx = lane; idx < 1024; idx += 32) {
    int r = idx >> 6, n = idx & 63;
    float v = sh[wave][r][n];
    if (flags & 1) v = silu_f(v);
    if (flags & 2) v = (v - smu[wave][r]) * srs[wave][r] * lng[n] + lnb[n];
    if (residual) v += (float)residual[(row0 + r) * 64 + n];
    Y[(row0 + r) * 64 + n] = (_Float16)v;
  }
}

// ---------------------------------------------------------------------------
// Fused edge message + scatter: per 16-edge tile,
//   m = LN(silu([edge_emb, send[s], rec[r]] @ W0 + b0) @ W1 + b1)
//   aggr[receiver] += m   (fp32 atomics)
// grid = (E/128, B).
// ---------------------------------------------------------------------------
__global__ __launch_bounds__(256) void edge_msg_kernel(
    const _Float16* __restrict__ edge_emb,
    const _Float16* __restrict__ send_rep, long sstride,
    const _Float16* __restrict__ rec_rep, long rstride,
    const int* __restrict__ senders, const int* __restrict__ receivers,
    const _Float16* __restrict__ W0p, const float* __restrict__ b0,
    const _Float16* __restrict__ W1p, const float* __restrict__ b1,
    const float* __restrict__ lng, const float* __restrict__ lnb,
    float* __restrict__ aggr, long astride) {
  __shared__ alignas(16) _Float16 hsh[8][16][64];
  __shared__ float sh[8][16][64];
  __shared__ float smu[8][16];
  __shared__ float srs[8][16];
  int lane = threadIdx.x & 31, wave = threadIdx.x >> 5, hi = lane >> 4;
  int b = blockIdx.y;
  const _Float16* send = send_rep + (long)b * sstride;
  const _Float16* rec = rec_rep + (long)b * rstride;
  float* ag = aggr + (long)b * astride;
  long tile = (long)blockIdx.x * 8 + wave;
  long e = tile * 16 + (lane & 15);
  const _Float16* rE = edge_emb + e * 64;
  const _Float16* rS = send + (long)senders[e] * 64;
  const _Float16* rR = rec + (long)receivers[e] * 64;
  v16h a[6];
  a[0] = a_frag_h(rE, 0, hi);  a[1] = a_frag_h(rE, 32, hi);
  a[2] = a_frag_h(rS, 0, hi);  a[3] = a_frag_h(rS, 32, hi);
  a[4] = a_frag_h(rR, 0, hi);  a[5] = a_frag_h(rR, 32, hi);
  int n0 = lane & 15;
#pragma unroll
  for (int nt = 0; nt < 4; ++nt) {
    v8f c = {};
#pragma unroll
    for (int kc = 0; kc < 6; ++kc)
      c = wmma16(a[kc], b_frag(W0p, kc * 4 + nt, lane), c);
    float bv = b0[nt * 16 + n0];
#pragma unroll
    for (int r = 0; r < 8; ++r)
      hsh[wave][r + 8 * hi][nt * 16 + n0] = (_Float16)silu_f(c[r] + bv);
  }
  __syncthreads();
  const _Float16* hrow = &hsh[wave][lane & 15][0];
  v16h h0 = a_frag_h(hrow, 0, hi);
  v16h h1 = a_frag_h(hrow, 32, hi);
#pragma unroll
  for (int nt = 0; nt < 4; ++nt) {
    v8f c = {};
    c = wmma16(h0, b_frag(W1p, nt, lane), c);
    c = wmma16(h1, b_frag(W1p, 4 + nt, lane), c);
    float bv = b1[nt * 16 + n0];
#pragma unroll
    for (int r = 0; r < 8; ++r) sh[wave][r + 8 * hi][nt * 16 + n0] = c[r] + bv;
  }
  __syncthreads();
  if (lane < 16) {
    float s = 0.f, s2 = 0.f;
    for (int n = 0; n < 64; ++n) { float v = sh[wave][lane][n]; s += v; s2 += v * v; }
    float mu = s * 0.015625f;
    smu[wave][lane] = mu;
    srs[wave][lane] = rsqrtf(fmaxf(s2 * 0.015625f - mu * mu, 0.f) + 1e-5f);
  }
  __syncthreads();
  for (int idx = lane; idx < 1024; idx += 32) {
    int r = idx >> 6, n = idx & 63;
    float v = (sh[wave][r][n] - smu[wave][r]) * srs[wave][r] * lng[n] + lnb[n];
    long recv = receivers[tile * 16 + r];
    atomicAdd(ag + recv * 64 + n, v);
  }
}

// ---------------------------------------------------------------------------
// Node update: out = rec + LN(silu([rec, aggr] @ W0 + b0) @ W1 + b1)
// K = 128 (2 kc from f16 rec, 2 kc from fp32 aggr). grid = (R/128, B).
// ---------------------------------------------------------------------------
__global__ __launch_bounds__(256) void aggr_update_kernel(
    const _Float16* __restrict__ rec_rep, long rstride,
    const float* __restrict__ aggr, long astride,
    const _Float16* __restrict__ W0p, const float* __restrict__ b0,
    const _Float16* __restrict__ W1p, const float* __restrict__ b1,
    const float* __restrict__ lng, const float* __restrict__ lnb,
    _Float16* __restrict__ out, long ostride) {
  __shared__ alignas(16) _Float16 hsh[8][16][64];
  __shared__ float sh[8][16][64];
  __shared__ float smu[8][16];
  __shared__ float srs[8][16];
  int lane = threadIdx.x & 31, wave = threadIdx.x >> 5, hi = lane >> 4;
  int b = blockIdx.y;
  const _Float16* rec = rec_rep + (long)b * rstride;
  const float* ag = aggr + (long)b * astride;
  long row0 = ((long)blockIdx.x * 8 + wave) * 16;
  const _Float16* rr = rec + (row0 + (lane & 15)) * 64;
  const float* ar = ag + (row0 + (lane & 15)) * 64;
  v16h a[4];
  a[0] = a_frag_h(rr, 0, hi);  a[1] = a_frag_h(rr, 32, hi);
  a[2] = a_frag_f(ar, 0, hi);  a[3] = a_frag_f(ar, 32, hi);
  int n0 = lane & 15;
#pragma unroll
  for (int nt = 0; nt < 4; ++nt) {
    v8f c = {};
#pragma unroll
    for (int kc = 0; kc < 4; ++kc)
      c = wmma16(a[kc], b_frag(W0p, kc * 4 + nt, lane), c);
    float bv = b0[nt * 16 + n0];
#pragma unroll
    for (int r = 0; r < 8; ++r)
      hsh[wave][r + 8 * hi][nt * 16 + n0] = (_Float16)silu_f(c[r] + bv);
  }
  __syncthreads();
  const _Float16* hrow = &hsh[wave][lane & 15][0];
  v16h h0 = a_frag_h(hrow, 0, hi);
  v16h h1 = a_frag_h(hrow, 32, hi);
#pragma unroll
  for (int nt = 0; nt < 4; ++nt) {
    v8f c = {};
    c = wmma16(h0, b_frag(W1p, nt, lane), c);
    c = wmma16(h1, b_frag(W1p, 4 + nt, lane), c);
    float bv = b1[nt * 16 + n0];
#pragma unroll
    for (int r = 0; r < 8; ++r) sh[wave][r + 8 * hi][nt * 16 + n0] = c[r] + bv;
  }
  __syncthreads();
  if (lane < 16) {
    float s = 0.f, s2 = 0.f;
    for (int n = 0; n < 64; ++n) { float v = sh[wave][lane][n]; s += v; s2 += v * v; }
    float mu = s * 0.015625f;
    smu[wave][lane] = mu;
    srs[wave][lane] = rsqrtf(fmaxf(s2 * 0.015625f - mu * mu, 0.f) + 1e-5f);
  }
  __syncthreads();
  _Float16* op = out + (long)b * ostride;
  for (int idx = lane; idx < 1024; idx += 32) {
    int r = idx >> 6, n = idx & 63;
    float v = (sh[wave][r][n] - smu[wave][r]) * srs[wave][r] * lng[n] + lnb[n];
    v += (float)rec[(row0 + r) * 64 + n];
    op[(row0 + r) * 64 + n] = (_Float16)v;
  }
}

// ---------------------------------------------------------------------------
// Output head: pred = silu(X @ W0 + b0) @ W1 + b1 (5 cols, padded to 16),
// written transposed to (B, 5, NGRID) fp32. grid = (NGRID/128, B).
// ---------------------------------------------------------------------------
__global__ __launch_bounds__(256) void output_kernel(
    const _Float16* __restrict__ X,
    const _Float16* __restrict__ W0p, const float* __restrict__ b0,
    const _Float16* __restrict__ W1p, const float* __restrict__ b1,
    float* __restrict__ out) {
  __shared__ alignas(16) _Float16 hsh[8][16][64];
  int lane = threadIdx.x & 31, wave = threadIdx.x >> 5, hi = lane >> 4;
  int b = blockIdx.y;
  long rowN0 = ((long)blockIdx.x * 8 + wave) * 16;
  const _Float16* rowp = X + ((long)b * NGRID + rowN0 + (lane & 15)) * 64;
  v16h a0 = a_frag_h(rowp, 0, hi);
  v16h a1 = a_frag_h(rowp, 32, hi);
  int n0 = lane & 15;
#pragma unroll
  for (int nt = 0; nt < 4; ++nt) {
    v8f c = {};
    c = wmma16(a0, b_frag(W0p, nt, lane), c);
    c = wmma16(a1, b_frag(W0p, 4 + nt, lane), c);
    float bv = b0[nt * 16 + n0];
#pragma unroll
    for (int r = 0; r < 8; ++r)
      hsh[wave][r + 8 * hi][nt * 16 + n0] = (_Float16)silu_f(c[r] + bv);
  }
  __syncthreads();
  const _Float16* hrow = &hsh[wave][lane & 15][0];
  v16h h0 = a_frag_h(hrow, 0, hi);
  v16h h1 = a_frag_h(hrow, 32, hi);
  v8f c = {};
  c = wmma16(h0, b_frag(W1p, 0, lane), c);
  c = wmma16(h1, b_frag(W1p, 1, lane), c);
  if (n0 < 5) {
    float bv = b1[n0];
#pragma unroll
    for (int r = 0; r < 8; ++r) {
      long node = rowN0 + r + 8 * hi;
      out[((long)b * 5 + n0) * NGRID + node] = c[r] + bv;
    }
  }
}

// ---------------------------------------------------------------------------
// Host side.
// Param leaf order: jax pytree flatten — dict keys sorted, so for each MLP:
// W0, W1, b0, b1, [ln_g, ln_b]; MLPs alphabetical:
// encoding_grid_mlp(1-6), g2m_aggr(7-12), g2m_edge(13-18), g2m_embedder(19-24),
// grid_embedder(25-30), m2g_aggr(31-36), m2g_edge(37-42), m2g_embedder(43-48),
// mesh_embedder(49-54), output_map(55-58). Then mesh_static(59), g2m_feat(60),
// m2g_feat(61), g2m_idx(62), m2g_idx(63).
// ---------------------------------------------------------------------------
extern "C" void kernel_launch(void* const* d_in, const int* in_sizes, int n_in,
                              void* d_out, int out_size, void* d_ws, size_t ws_size,
                              hipStream_t stream) {
  (void)in_sizes; (void)n_in; (void)out_size; (void)ws_size;
  const float* prev = (const float*)d_in[0];
  auto F = [&](int i) { return (const float*)d_in[i]; };
  const int* g2m_idx = (const int*)d_in[62];
  const int* m2g_idx = (const int*)d_in[63];
  float* outp = (float*)d_out;

  // ---- workspace carving ----
  char* ws = (char*)d_ws;
  size_t off = 0;
  auto carve = [&](size_t bytes) -> char* {
    char* p = ws + off;
    off += (bytes + 255) & ~(size_t)255;
    return p;
  };
  _Float16* bufA     = (_Float16*)carve((size_t)NB * NGRID * 64 * 2); // hidden scratch / final grid rep
  _Float16* grid_emb = (_Float16*)carve((size_t)NB * NGRID * 64 * 2);
  _Float16* grid_rep = (_Float16*)carve((size_t)NB * NGRID * 64 * 2);
  _Float16* mesh_emb = (_Float16*)carve((size_t)NMESH * 64 * 2);
  _Float16* mesh_rep = (_Float16*)carve((size_t)NB * NMESH * 64 * 2);
  _Float16* edge_emb = (_Float16*)carve((size_t)NEDGE * 64 * 2);
  _Float16* edge_h   = (_Float16*)carve((size_t)NEDGE * 64 * 2);
  float* aggr_mesh   = (float*)carve((size_t)NB * NMESH * 64 * 4);
  float* aggr_grid   = (float*)carve((size_t)NB * NGRID * 64 * 4);
  auto wbuf = [&](int K) { return (_Float16*)carve((size_t)K * 128); };
  _Float16* wp_grid1 = wbuf(64);
  _Float16* wp_mesh1 = wbuf(64);
  _Float16* wp_g2me1 = wbuf(64);
  _Float16* wp_m2ge1 = wbuf(64);
  _Float16* wp_enc0  = wbuf(64);
  _Float16* wp_enc1  = wbuf(64);
  _Float16* wp_ge0   = wbuf(192);
  _Float16* wp_ge1   = wbuf(64);
  _Float16* wp_ga0   = wbuf(128);
  _Float16* wp_ga1   = wbuf(64);
  _Float16* wp_me0   = wbuf(192);
  _Float16* wp_me1   = wbuf(64);
  _Float16* wp_ma0   = wbuf(128);
  _Float16* wp_ma1   = wbuf(64);
  _Float16* wp_o0    = wbuf(64);
  _Float16* wp_o1    = (_Float16*)carve((size_t)64 * 16 * 2); // 64x5 padded to 16 cols

  // ---- zero scatter accumulators ----
  hipMemsetAsync(aggr_mesh, 0, (size_t)NB * NMESH * 64 * 4, stream);
  hipMemsetAsync(aggr_grid, 0, (size_t)NB * NGRID * 64 * 4, stream);

  // ---- pack weights to WMMA B-fragment layout ----
  auto pack = [&](const float* W, _Float16* dst, int K, int Ncols) {
    int NT = (Ncols + 15) >> 4;
    int total = (K >> 5) * NT * 512;
    pack_w_kernel<<<(total + 255) / 256, 256, 0, stream>>>(W, dst, K, Ncols);
  };
  pack(F(26), wp_grid1, 64, 64);   // grid_embedder W1
  pack(F(50), wp_mesh1, 64, 64);   // mesh_embedder W1
  pack(F(20), wp_g2me1, 64, 64);   // g2m_embedder W1
  pack(F(44), wp_m2ge1, 64, 64);   // m2g_embedder W1
  pack(F(1),  wp_enc0,  64, 64);   // encoding_grid_mlp W0
  pack(F(2),  wp_enc1,  64, 64);   // encoding_grid_mlp W1
  pack(F(13), wp_ge0,  192, 64);   // g2m_edge W0
  pack(F(14), wp_ge1,   64, 64);   // g2m_edge W1
  pack(F(7),  wp_ga0,  128, 64);   // g2m_aggr W0
  pack(F(8),  wp_ga1,   64, 64);   // g2m_aggr W1
  pack(F(37), wp_me0,  192, 64);   // m2g_edge W0
  pack(F(38), wp_me1,   64, 64);   // m2g_edge W1
  pack(F(31), wp_ma0,  128, 64);   // m2g_aggr W0
  pack(F(32), wp_ma1,   64, 64);   // m2g_aggr W1
  pack(F(55), wp_o0,    64, 64);   // output W0
  pack(F(56), wp_o1,    64, 5);    // output W1 (64x5, padded)

  const long GS = (long)NGRID * 64;   // per-batch grid rep stride (elements)
  const long MS = (long)NMESH * 64;

  // ---- grid embedder ----
  grid_l1_kernel<<<(NB * NGRID * 64) / 256, 256, 0, stream>>>(prev, F(25), F(27), bufA);
  gemm64_kernel<<<(NB * NGRID) / 128, 256, 0, stream>>>(
      bufA, wp_grid1, F(28), F(29), F(30), nullptr, grid_emb, 2);

  // ---- mesh embedder ----
  feat3_l1_kernel<<<(NMESH * 64) / 256, 256, 0, stream>>>(F(59), F(49), F(51), edge_h);
  gemm64_kernel<<<NMESH / 128, 256, 0, stream>>>(
      edge_h, wp_mesh1, F(52), F(53), F(54), nullptr, mesh_emb, 2);

  // ---- g2m edge embedder ----
  feat3_l1_kernel<<<(NEDGE * 64) / 256, 256, 0, stream>>>(F(60), F(19), F(21), edge_h);
  gemm64_kernel<<<NEDGE / 128, 256, 0, stream>>>(
      edge_h, wp_g2me1, F(22), F(23), F(24), nullptr, edge_emb, 2);

  // ---- g2m interaction: messages + scatter ----
  edge_msg_kernel<<<dim3(NEDGE / 128, NB), 256, 0, stream>>>(
      edge_emb, grid_emb, GS, mesh_emb, 0, g2m_idx, g2m_idx + NEDGE,
      wp_ge0, F(15), wp_ge1, F(16), F(17), F(18), aggr_mesh, MS);
  // ---- mesh node update -> mesh_rep ----
  aggr_update_kernel<<<dim3(NMESH / 128, NB), 256, 0, stream>>>(
      mesh_emb, 0, aggr_mesh, MS, wp_ga0, F(9), wp_ga1, F(10), F(11), F(12),
      mesh_rep, MS);

  // ---- encoding grid mlp: grid_rep = grid_emb + mlp(grid_emb) ----
  gemm64_kernel<<<(NB * NGRID) / 128, 256, 0, stream>>>(
      grid_emb, wp_enc0, F(3), nullptr, nullptr, nullptr, bufA, 1);
  gemm64_kernel<<<(NB * NGRID) / 128, 256, 0, stream>>>(
      bufA, wp_enc1, F(4), F(5), F(6), grid_emb, grid_rep, 2);

  // ---- m2g edge embedder (reuses edge buffers) ----
  feat3_l1_kernel<<<(NEDGE * 64) / 256, 256, 0, stream>>>(F(61), F(43), F(45), edge_h);
  gemm64_kernel<<<NEDGE / 128, 256, 0, stream>>>(
      edge_h, wp_m2ge1, F(46), F(47), F(48), nullptr, edge_emb, 2);

  // ---- m2g interaction ----
  edge_msg_kernel<<<dim3(NEDGE / 128, NB), 256, 0, stream>>>(
      edge_emb, mesh_rep, MS, grid_rep, GS, m2g_idx, m2g_idx + NEDGE,
      wp_me0, F(39), wp_me1, F(40), F(41), F(42), aggr_grid, GS);
  // ---- grid node update -> bufA (final grid rep) ----
  aggr_update_kernel<<<dim3(NGRID / 128, NB), 256, 0, stream>>>(
      grid_rep, GS, aggr_grid, GS, wp_ma0, F(33), wp_ma1, F(34), F(35), F(36),
      bufA, GS);

  // ---- output head -> d_out (B, 5, H, W) ----
  output_kernel<<<dim3(NGRID / 128, NB), 256, 0, stream>>>(
      bufA, wp_o0, F(57), wp_o1, F(58), outp);
}
